// RegionGATLayer_81355270521335
// MI455X (gfx1250) — matrix operand order repaired and checked by reading, not compiled, not run on hardware
//
#include <hip/hip_runtime.h>
#include <hip/hip_bf16.h>

// RegionGATLayer fused flash-GAT for MI455X (gfx1250, wave32, WMMA).
// B=4, N=2048, F=128, H=4, D=32.

typedef __attribute__((ext_vector_type(16))) __bf16 v16bf;
typedef __attribute__((ext_vector_type(8)))  __bf16 v8bf;
typedef __attribute__((ext_vector_type(8)))  float  v8f;
typedef __attribute__((ext_vector_type(4)))  float  f4;

#define B_  4
#define N_  2048
#define F_  128
#define H_  4
#define D_  32
#define NEG_INF_ (-1.0e9f)
#define SLOPE_   0.2f

// Async global->LDS (ASYNCcnt-tracked) path, if this toolchain exposes it.
#if defined(__has_builtin)
#if __has_builtin(__builtin_amdgcn_global_load_async_to_lds_b128)
#define HAVE_ASYNC_LDS 1
#endif
#endif

#ifdef HAVE_ASYNC_LDS
// Param types per hipcc diagnostic: pointee is a 16B int vector; addrspace(1)
// prints as __device__, LDS side is addrspace(3).  Generic->AS casts go via
// integer bit-cast; LDS generic addresses carry the LDS byte offset in the
// low 32 bits (ISA 10.2 aperture rules).
typedef int vs4i __attribute__((vector_size(16)));
typedef __attribute__((address_space(1))) vs4i g_vs4i;
typedef __attribute__((address_space(3))) vs4i l_vs4i;
#define AS1(p) ((g_vs4i*)(uintptr_t)(p))
#define AS3(p) ((l_vs4i*)(unsigned)(uintptr_t)(p))

__device__ __forceinline__ void wait_async0() {
#if __has_builtin(__builtin_amdgcn_s_wait_asynccnt)
  __builtin_amdgcn_s_wait_asynccnt(0);
#else
  asm volatile("s_wait_asynccnt 0" ::: "memory");
#endif
}
#endif

// ---------------------------------------------------------------------------
// Kernel 1: h = x @ W per head, stored transposed as hT[b*H+h][d][n] in bf16.
// One wave computes a 16-node x 32-dim tile, K=128 in 4 steps of
// v_wmma_f32_16x16x32_bf16.  W (per head) staged transposed in LDS as bf16.
// grid = (B*H, N/128), block = 256 (8 waves).
// ---------------------------------------------------------------------------
__global__ __launch_bounds__(256) void gat_project(
    const float* __restrict__ x, const float* __restrict__ W,
    __bf16* __restrict__ hT)
{
  __shared__ __align__(16) __bf16 sWT[D_ * F_];   // [d][f], 8 KB
  const int bh = blockIdx.x;
  const int b  = bh / H_;
  const int hd = bh % H_;

  for (int i = threadIdx.x; i < D_ * F_; i += 256) {
    int d = i / F_, f = i % F_;
    sWT[i] = (__bf16)W[((size_t)hd * F_ + f) * D_ + d];
  }
  __syncthreads();

  const int wave = threadIdx.x >> 5;
  const int lane = threadIdx.x & 31;
  const int l15  = lane & 15;
  const int hi   = lane >> 4;
  const int nodebase = blockIdx.y * 128 + wave * 16;

  const float* xrow = x + ((size_t)b * N_ + nodebase + l15) * F_;

  v8f acc0 = {}; v8f acc1 = {};
  #pragma unroll
  for (int kt = 0; kt < 4; ++kt) {
    // A operand (x tile, 16x32): lane row = l15, K = (e&7) + 16*(e>>3) + 8*hi
    const int f0 = kt * 32 + 8 * hi;
    f4 xa = *(const f4*)(xrow + f0);
    f4 xb = *(const f4*)(xrow + f0 + 4);
    f4 xc = *(const f4*)(xrow + f0 + 16);
    f4 xd = *(const f4*)(xrow + f0 + 20);
    v16bf a;
    #pragma unroll
    for (int e = 0; e < 4; ++e) {
      a[e]      = (__bf16)xa[e];
      a[e + 4]  = (__bf16)xb[e];
      a[e + 8]  = (__bf16)xc[e];
      a[e + 12] = (__bf16)xd[e];
    }
    // B operand (W tile, 32x16 per column tile): N = l15, K = (e&7) + 16*(e>>3) + 8*hi
    const int fB = kt * 32 + 8 * hi;
    const __bf16* w0 = &sWT[(0  + l15) * F_ + fB];
    const __bf16* w1 = &sWT[(16 + l15) * F_ + fB];
    v8bf t0 = *(const v8bf*)w0;  v8bf t1 = *(const v8bf*)(w0 + 16);
    v8bf u0 = *(const v8bf*)w1;  v8bf u1 = *(const v8bf*)(w1 + 16);
    v16bf b0, b1;
    #pragma unroll
    for (int e = 0; e < 8; ++e) {
      b0[e] = t0[e]; b0[e + 8] = t1[e];
      b1[e] = u0[e]; b1[e + 8] = u1[e];
    }
    acc0 = __builtin_amdgcn_wmma_f32_16x16x32_bf16(false, a, false, b0, (short)0, acc0, false, false);
    acc1 = __builtin_amdgcn_wmma_f32_16x16x32_bf16(false, a, false, b1, (short)0, acc1, false, false);
  }

  // C layout: VGPR r -> node M = r + 8*hi, col N = l15.  Pack 8 consecutive
  // nodes per lane -> one 16B store into hT[d][n].
  #pragma unroll
  for (int c = 0; c < 2; ++c) {
    v8f acc = c ? acc1 : acc0;
    v8bf pkt;
    #pragma unroll
    for (int r = 0; r < 8; ++r) pkt[r] = (__bf16)acc[r];
    int d = c * 16 + l15;
    size_t off = ((size_t)bh * D_ + d) * N_ + nodebase + 8 * hi;
    *(v8bf*)(hT + off) = pkt;
  }
}

// ---------------------------------------------------------------------------
// Kernel 2: al[n] = h[n,:]·a_left, ar[n] = h[n,:]·a_right (coalesced over n).
// ---------------------------------------------------------------------------
__global__ __launch_bounds__(256) void gat_coef(
    const __bf16* __restrict__ hT, const float* __restrict__ aL,
    const float* __restrict__ aR, float* __restrict__ al, float* __restrict__ ar)
{
  int t  = blockIdx.x * 256 + threadIdx.x;      // 0 .. B*H*N-1
  int bh = t / N_;
  int n  = t % N_;
  int hd = bh % H_;
  float sl = 0.f, sr = 0.f;
  #pragma unroll
  for (int d = 0; d < D_; ++d) {
    float hv = (float)hT[((size_t)bh * D_ + d) * N_ + n];
    sl += hv * aL[hd * D_ + d];
    sr += hv * aR[hd * D_ + d];
  }
  al[t] = sl;
  ar[t] = sr;
}

// ---------------------------------------------------------------------------
// Kernel 3: fused masked-softmax aggregation (flash style, no (N,N) spill).
// One block per (b,h, 128-row slab); the full hT plane (128 KB bf16) + ar
// (8 KB) staged into LDS (320 KB/WGP) via async global->LDS DMA when the
// toolchain exposes it (ASYNCcnt path), else via b128 register copies.
// Each wave owns 16 rows, streams 64 column tiles of 32 with online softmax
// and wmma P·h accumulation; adj stays resident in the 192 MB L2.
// ---------------------------------------------------------------------------
__global__ __launch_bounds__(256) void gat_attn(
    const __bf16* __restrict__ hT, const float* __restrict__ adj,
    const float* __restrict__ al_g, const float* __restrict__ ar_g,
    const float* __restrict__ bias, float* __restrict__ out)
{
  extern __shared__ __align__(16) char smem[];
  __bf16* sH  = (__bf16*)smem;                   // [32][2048] bf16, 128 KB
  float*  sAr = (float*)(smem + (size_t)D_ * N_ * 2);  // 8 KB

  const int bh = blockIdx.x;
  const int b  = bh / H_;
  const int hd = bh % H_;

  {
    const v8bf* src = (const v8bf*)(hT + (size_t)bh * D_ * N_);
    v8bf* dst = (v8bf*)sH;
    const f4* asrc = (const f4*)(ar_g + bh * N_);
    f4* adst = (f4*)sAr;
#ifdef HAVE_ASYNC_LDS
    // ASYNCcnt-tracked DMA: 16B per lane per issue, no VGPR round-trip.
    for (int i = threadIdx.x; i < (D_ * N_) / 8; i += 256)
      __builtin_amdgcn_global_load_async_to_lds_b128(AS1(src + i), AS3(dst + i), 0, 0);
    for (int i = threadIdx.x; i < N_ / 4; i += 256)
      __builtin_amdgcn_global_load_async_to_lds_b128(AS1(asrc + i), AS3(adst + i), 0, 0);
    wait_async0();
#else
    for (int i = threadIdx.x; i < (D_ * N_) / 8; i += 256) dst[i] = src[i];
    for (int i = threadIdx.x; i < N_ / 4; i += 256) adst[i] = asrc[i];
#endif
  }
  __syncthreads();

  const int wave = threadIdx.x >> 5;
  const int lane = threadIdx.x & 31;
  const int l15  = lane & 15;
  const int hi   = lane >> 4;
  const int rowbase = blockIdx.y * 128 + wave * 16;
  const int row     = rowbase + l15;

  const float  alv    = al_g[bh * N_ + row];
  const float* adjrow = adj + (size_t)row * N_;

  float m_run = -3.0e38f, l_run = 0.f;
  v8f acc0 = {}; v8f acc1 = {};

  for (int jt = 0; jt < N_ / 32; ++jt) {
    const int j0 = jt * 32;
    if (jt + 1 < N_ / 32)
      __builtin_prefetch(adjrow + j0 + 32, 0, 1);   // global_prefetch_b8

    // scores in A-operand layout: lane row = l15, K = (e&7)+16*(e>>3)+8*hi
    const int ja = j0 + 8 * hi;
    f4 A0 = *(const f4*)(adjrow + ja);
    f4 A1 = *(const f4*)(adjrow + ja + 4);
    f4 A2 = *(const f4*)(adjrow + ja + 16);
    f4 A3 = *(const f4*)(adjrow + ja + 20);
    f4 R0 = *(const f4*)(sAr + ja);
    f4 R1 = *(const f4*)(sAr + ja + 4);
    f4 R2 = *(const f4*)(sAr + ja + 16);
    f4 R3 = *(const f4*)(sAr + ja + 20);

    float ev[16];
    #pragma unroll
    for (int e = 0; e < 4; ++e) {
      float e0 = alv + R0[e]; e0 = e0 >= 0.f ? e0 : SLOPE_ * e0; ev[e]      = A0[e] > 0.5f ? e0 : NEG_INF_;
      float e1 = alv + R1[e]; e1 = e1 >= 0.f ? e1 : SLOPE_ * e1; ev[e + 4]  = A1[e] > 0.5f ? e1 : NEG_INF_;
      float e2 = alv + R2[e]; e2 = e2 >= 0.f ? e2 : SLOPE_ * e2; ev[e + 8]  = A2[e] > 0.5f ? e2 : NEG_INF_;
      float e3 = alv + R3[e]; e3 = e3 >= 0.f ? e3 : SLOPE_ * e3; ev[e + 12] = A3[e] > 0.5f ? e3 : NEG_INF_;
    }

    // online softmax: row max across the lane pair (l15, l15+16)
    float mloc = ev[0];
    #pragma unroll
    for (int e = 1; e < 16; ++e) mloc = fmaxf(mloc, ev[e]);
    mloc = fmaxf(mloc, __shfl_xor(mloc, 16, 32));
    float m_new = fmaxf(m_run, mloc);
    float scale = __expf(m_run - m_new);

    v16bf pa; float psum = 0.f;
    #pragma unroll
    for (int e = 0; e < 16; ++e) {
      float p = __expf(ev[e] - m_new);
      psum += p;
      pa[e] = (__bf16)p;
    }
    psum += __shfl_xor(psum, 16, 32);
    l_run = l_run * scale + psum;
    m_run = m_new;

    // rescale accumulator rows (C row M = r + 8*hi lives in lane r+8*hi)
    #pragma unroll
    for (int r = 0; r < 8; ++r) {
      float s = __shfl(scale, r + 8 * hi, 32);
      acc0[r] *= s; acc1[r] *= s;
    }

    // B operand from LDS hT[d][j]: N = l15, K = (e&7)+16*(e>>3)+8*hi -> two 16B loads
    const __bf16* p0 = sH + (size_t)(0  + l15) * N_ + j0 + 8 * hi;
    const __bf16* p1 = sH + (size_t)(16 + l15) * N_ + j0 + 8 * hi;
    v8bf t0 = *(const v8bf*)p0; v8bf t1 = *(const v8bf*)(p0 + 16);
    v8bf u0 = *(const v8bf*)p1; v8bf u1 = *(const v8bf*)(p1 + 16);
    v16bf b0, b1;
    #pragma unroll
    for (int e = 0; e < 8; ++e) {
      b0[e] = t0[e]; b0[e + 8] = t1[e];
      b1[e] = u0[e]; b1[e + 8] = u1[e];
    }
    acc0 = __builtin_amdgcn_wmma_f32_16x16x32_bf16(false, pa, false, b0, (short)0, acc0, false, false);
    acc1 = __builtin_amdgcn_wmma_f32_16x16x32_bf16(false, pa, false, b1, (short)0, acc1, false, false);
  }

  // epilogue: normalize, bias, ELU, scatter to (B, N, H*D)
  #pragma unroll
  for (int c = 0; c < 2; ++c) {
    v8f acc = c ? acc1 : acc0;
    int d = c * 16 + l15;
    float bv = bias[hd * D_ + d];
    #pragma unroll
    for (int r = 0; r < 8; ++r) {
      float lr = __shfl(l_run, r + 8 * hi, 32);
      float o  = acc[r] / lr + bv;
      o = o >= 0.f ? o : (__expf(o) - 1.0f);
      int node = rowbase + r + 8 * hi;
      out[((size_t)b * N_ + node) * (H_ * D_) + hd * D_ + d] = o;
    }
  }
}

// ---------------------------------------------------------------------------
extern "C" void kernel_launch(void* const* d_in, const int* in_sizes, int n_in,
                              void* d_out, int out_size, void* d_ws, size_t ws_size,
                              hipStream_t stream) {
  const float* x    = (const float*)d_in[0];
  const float* adj  = (const float*)d_in[1];
  const float* W    = (const float*)d_in[2];
  const float* aL   = (const float*)d_in[3];
  const float* aR   = (const float*)d_in[4];
  const float* bias = (const float*)d_in[5];
  float* out = (float*)d_out;

  char* ws = (char*)d_ws;
  __bf16* hT = (__bf16*)ws;                               // 2 MB
  float*  al = (float*)(ws + 2u * 1024u * 1024u);         // 128 KB
  float*  ar = (float*)(ws + 2u * 1024u * 1024u + 128u * 1024u);

  dim3 blk(256);
  gat_project<<<dim3(B_ * H_, N_ / 128), blk, 0, stream>>>(x, W, hT);
  gat_coef<<<dim3((B_ * H_ * N_) / 256), blk, 0, stream>>>(hT, aL, aR, al, ar);
  size_t shmem = (size_t)D_ * N_ * 2 + (size_t)N_ * 4;    // 136 KB (< 320 KB/WGP)
  gat_attn<<<dim3(B_ * H_, N_ / 128), blk, shmem, stream>>>(hT, adj, al, ar, bias, out);
}
